// CSA_43714177138971
// MI455X (gfx1250) — matrix operand order repaired and checked by reading, not compiled, not run on hardware
//
#include <hip/hip_runtime.h>
#include <stdint.h>

// ---------- problem constants ----------
#define NB     8
#define NC     384
#define NH     64
#define NW     64
#define PHW    32          // pooled h == pooled w
#define NL     1024        // PHW*PHW
#define NBL    8192        // NB*NL
#define NK4H   972         // K^4 * HEADS
#define NKK    9
#define NGROUPS (NBL * 12 * 9)   // softmax groups of 9

// GEMM tile geometry: 128x64 block tile, K-step 32, double-buffered LDS
#define TASZ   (128 * 40)  // ushorts per A buffer (row stride 40 = 32 + 8 pad)
#define TBSZ   (64 * 40)   // ushorts per B buffer

// ---------- CDNA5 async global->LDS (guarded: falls back to sync staging) ----------
#if defined(__gfx1250__) && __has_builtin(__builtin_amdgcn_global_load_async_to_lds_b128) && \
    __has_builtin(__builtin_amdgcn_s_wait_asynccnt)
#define USE_ASYNC_LDS 1
#else
#define USE_ASYNC_LDS 0
#endif

// ---------- WMMA types ----------
typedef __attribute__((ext_vector_type(16))) __bf16        v16bf;
typedef __attribute__((ext_vector_type(8)))  float         v8f;
typedef __attribute__((ext_vector_type(4)))  unsigned int  u32x4;
typedef __attribute__((ext_vector_type(4)))  int           v4i;

union Frag16 { v16bf bf; u32x4 q[2]; unsigned short s[16]; };
union H8     { u32x4 q;  unsigned short s[8]; };

__device__ __forceinline__ unsigned short f2bf(float f) {
  unsigned int u = __float_as_uint(f);
  u += 0x7FFFu + ((u >> 16) & 1u);            // round-to-nearest-even
  return (unsigned short)(u >> 16);
}
__device__ __forceinline__ float bf2f(unsigned short s) {
  return __uint_as_float(((unsigned int)s) << 16);
}

#if USE_ASYNC_LDS
// copies 16 bytes global->LDS; OFF (bytes, ICE) is applied to BOTH addresses by HW
template <int OFF>
__device__ __forceinline__ void async_copy16(const unsigned short* g, unsigned short* l) {
  __builtin_amdgcn_global_load_async_to_lds_b128(
      (v4i __attribute__((address_space(1)))*)g,
      (v4i __attribute__((address_space(3)))*)l, OFF, 0);
}
#endif

__device__ __forceinline__ void gemm_sync() {
#if USE_ASYNC_LDS
  __builtin_amdgcn_s_wait_asynccnt(0);
#endif
  __syncthreads();
}

// A fragment (16x32 bf16), ISA 7.12.2: lanes0-15 row M=lane, elems = K0..7,K16..23
//                                      lanes16-31 row M=lane-16, elems = K8..15,K24..31
__device__ __forceinline__ v16bf load_a_frag(const unsigned short* sA, int rowBase, int lane) {
  const int lr   = lane & 15;
  const int koff = (lane < 16) ? 0 : 8;
  const unsigned short* p = sA + (rowBase + lr) * 40 + koff;
  Frag16 f;
  f.q[0] = *(const u32x4*)(p);        // K = koff .. koff+7
  f.q[1] = *(const u32x4*)(p + 16);   // K = koff+16 .. koff+23
  return f.bf;
}
// B fragment (32x16 bf16), stored [N][K] in LDS: lanes0-15 K=0..7,K16..23, lanes16-31 K=8..15,K24..31, N=lane%16
__device__ __forceinline__ v16bf load_b_frag(const unsigned short* sB, int colBase, int lane) {
  const int ln   = lane & 15;
  const int koff = (lane < 16) ? 0 : 8;
  const unsigned short* p = sB + (colBase + ln) * 40 + koff;
  Frag16 f;
  f.q[0] = *(const u32x4*)(p);
  f.q[1] = *(const u32x4*)(p + 16);
  return f.bf;
}

__device__ __forceinline__ v8f wmma_bf16(v16bf a, v16bf b, v8f c) {
#if defined(__gfx1250__)
  return __builtin_amdgcn_wmma_f32_16x16x32_bf16(false, a, false, b, (short)0, c, false, false);
#else
  return c;
#endif
}

// Double-buffered main loop: prefetch tile k+1 into alternate buffer while
// computing tile k; one asynccnt-wait + one workgroup barrier per K-step.
template <typename ALoad, typename BLoad>
__device__ __forceinline__ void gemm_mainloop(const unsigned short* sA, const unsigned short* sB,
                                              ALoad aload, BLoad bload, int lane, int wm, int wn,
                                              v8f& acc00, v8f& acc01, v8f& acc10, v8f& acc11) {
  aload(0, 0);
  bload(0, 0);
  gemm_sync();
  for (int kt = 0; kt < NC / 32; ++kt) {
    const int cur = kt & 1;
    if (kt + 1 < NC / 32) {
      aload((kt + 1) * 32, cur ^ 1);
      bload((kt + 1) * 32, cur ^ 1);
    }
    const unsigned short* cA = sA + cur * TASZ;
    const unsigned short* cB = sB + cur * TBSZ;
    v16bf a0 = load_a_frag(cA, wm, lane);
    v16bf a1 = load_a_frag(cA, wm + 16, lane);
    v16bf b0 = load_b_frag(cB, wn, lane);
    v16bf b1 = load_b_frag(cB, wn + 16, lane);
    acc00 = wmma_bf16(a0, b0, acc00);
    acc01 = wmma_bf16(a0, b1, acc01);
    acc10 = wmma_bf16(a1, b0, acc10);
    acc11 = wmma_bf16(a1, b1, acc11);
    gemm_sync();
  }
}

// =====================================================================
// fp32 -> bf16 weight conversion
// =====================================================================
__global__ void cvt_bf16_kernel(const float* __restrict__ s, unsigned short* __restrict__ d, int n) {
  for (int i = blockIdx.x * blockDim.x + threadIdx.x; i < n; i += gridDim.x * blockDim.x)
    d[i] = f2bf(s[i]);
}

// =====================================================================
// 2x2 average pool, output row-major [bl][c] bf16
// =====================================================================
__global__ __launch_bounds__(256) void pool_kernel(const float* __restrict__ x,
                                                   unsigned short* __restrict__ poolbf) {
  const int blk = blockIdx.x;          // NB * PHW blocks
  const int b = blk >> 5;
  const int i = blk & 31;
  for (int idx = threadIdx.x; idx < NC * PHW; idx += 256) {
    const int c = idx >> 5;
    const int j = idx & 31;
    const float* px = x + (((size_t)(b * NC + c) * NH + 2 * i) * NW + 2 * j);
    const float s = 0.25f * (px[0] + px[1] + px[NW] + px[NW + 1]);
    poolbf[(size_t)(b * NL + i * PHW + j) * NC + c] = f2bf(s);
  }
}

// =====================================================================
// NCHW fp32 -> NHWC bf16 transpose (so im2col rows are contiguous)
// =====================================================================
__global__ __launch_bounds__(256) void transpose_kernel(const float* __restrict__ x,
                                                        unsigned short* __restrict__ xT) {
  __shared__ float tile[64][65];
  const int bh = blockIdx.x;
  const int b  = bh >> 6;
  const int hh = bh & 63;
  const int c0 = blockIdx.y * 64;
  for (int idx = threadIdx.x; idx < 64 * 64; idx += 256) {
    const int c  = idx >> 6;
    const int ww = idx & 63;
    tile[c][ww] = x[((size_t)(b * NC + c0 + c) * NH + hh) * NW + ww];
  }
  __syncthreads();
  for (int idx = threadIdx.x; idx < 64 * 64; idx += 256) {
    const int ww = idx >> 6;
    const int c  = idx & 63;
    xT[((size_t)(b * NH + hh) * NW + ww) * NC + c0 + c] = f2bf(tile[c][ww]);
  }
}

// =====================================================================
// attn GEMM: [8192,384] x [384,972]  (B given as [N,K] row-major), +bias, fp32 out
// =====================================================================
__global__ __launch_bounds__(256) void attn_gemm_kernel(const unsigned short* __restrict__ Apool,
                                                        const unsigned short* __restrict__ Wb,
                                                        const float* __restrict__ attn_b,
                                                        float* __restrict__ alog) {
  __shared__ __align__(16) unsigned short sA[2 * TASZ];
  __shared__ __align__(16) unsigned short sB[2 * TBSZ];
  const int tid  = threadIdx.x;
  const int lane = tid & 31;
  const int wave = tid >> 5;
  const int wm = (wave & 3) * 32;
  const int wn = (wave >> 2) * 32;
  const int m0 = blockIdx.x * 128;
  const int n0 = blockIdx.y * 64;
  const int ar = tid >> 1, ah = (tid & 1) * 16;
  const int br = tid >> 2, bq = (tid & 3) * 8;
  const int gbn = n0 + br;
  const bool bvalid = gbn < NK4H;
  const unsigned short* arow = Apool + (size_t)(m0 + ar) * NC + ah;
  const unsigned short* brow = Wb + (size_t)(bvalid ? gbn : 0) * NC + bq;
  if (!bvalid) {   // pre-zero both buffers' pad rows once; loaders skip them
    u32x4 z = {0u, 0u, 0u, 0u};
    *(u32x4*)&sB[0 * TBSZ + br * 40 + bq] = z;
    *(u32x4*)&sB[1 * TBSZ + br * 40 + bq] = z;
  }
  v8f acc00 = {}, acc01 = {}, acc10 = {}, acc11 = {};
  auto aload = [&](int k0, int buf) {
    unsigned short* dst = &sA[buf * TASZ + ar * 40 + ah];
#if USE_ASYNC_LDS
    async_copy16<0>(arow + k0, dst);
    async_copy16<16>(arow + k0, dst);
#else
    *(u32x4*)dst       = *(const u32x4*)(arow + k0);
    *(u32x4*)(dst + 8) = *(const u32x4*)(arow + k0 + 8);
    __builtin_prefetch((const void*)(arow + k0 + 32), 0, 3);
#endif
  };
  auto bload = [&](int k0, int buf) {
    if (!bvalid) return;
    unsigned short* dst = &sB[buf * TBSZ + br * 40 + bq];
#if USE_ASYNC_LDS
    async_copy16<0>(brow + k0, dst);
#else
    *(u32x4*)dst = *(const u32x4*)(brow + k0);
    __builtin_prefetch((const void*)(brow + k0 + 32), 0, 3);
#endif
  };
  gemm_mainloop(sA, sB, aload, bload, lane, wm, wn, acc00, acc01, acc10, acc11);

  const int ln = lane & 15;
  const int rb = (lane < 16) ? 0 : 8;
  const int col0 = n0 + wn + ln;
  const int col1 = col0 + 16;
  const float bi0 = (col0 < NK4H) ? attn_b[col0] : 0.f;
  const float bi1 = (col1 < NK4H) ? attn_b[col1] : 0.f;
#pragma unroll
  for (int r = 0; r < 8; ++r) {
    const int row0 = m0 + wm + rb + r;
    const int row1 = row0 + 16;
    if (col0 < NK4H) {
      alog[(size_t)row0 * NK4H + col0] = acc00[r] + bi0;
      alog[(size_t)row1 * NK4H + col0] = acc10[r] + bi0;
    }
    if (col1 < NK4H) {
      alog[(size_t)row0 * NK4H + col1] = acc01[r] + bi1;
      alog[(size_t)row1 * NK4H + col1] = acc11[r] + bi1;
    }
  }
}

// =====================================================================
// softmax over contiguous groups of 9 (scale applied here), in-place
// =====================================================================
__global__ void softmax9_kernel(float* __restrict__ a, int ngroups) {
  const int g = blockIdx.x * blockDim.x + threadIdx.x;
  if (g >= ngroups) return;
  float* p = a + (size_t)g * 9;
  const float scale = 0.17677669529663687f;   // 32^-0.5
  float v[9];
  float m = -3.4e38f;
#pragma unroll
  for (int q = 0; q < 9; ++q) { v[q] = p[q] * scale; m = fmaxf(m, v[q]); }
  float s = 0.f;
#pragma unroll
  for (int q = 0; q < 9; ++q) { v[q] = __expf(v[q] - m); s += v[q]; }
  const float inv = 1.f / s;
#pragma unroll
  for (int q = 0; q < 9; ++q) p[q] = v[q] * inv;
}

// =====================================================================
// conv GEMM (per tap kk): A = on-the-fly im2col from xT (NHWC bf16),
// B = conv_w[kk] ([N=o, K=c] row-major), out v_bf[bl][kk][o] bf16
// =====================================================================
__global__ __launch_bounds__(256) void conv_gemm_kernel(const unsigned short* __restrict__ xT,
                                                        const unsigned short* __restrict__ Wconv,
                                                        unsigned short* __restrict__ vbf) {
  __shared__ __align__(16) unsigned short sA[2 * TASZ];
  __shared__ __align__(16) unsigned short sB[2 * TBSZ];
  const int tid  = threadIdx.x;
  const int lane = tid & 31;
  const int wave = tid >> 5;
  const int wm = (wave & 3) * 32;
  const int wn = (wave >> 2) * 32;
  const int m0 = blockIdx.x * 128;
  const int n0 = blockIdx.y * 64;
  const int kk = blockIdx.z;
  const int ki = kk / 3, kj = kk - ki * 3;
  const int ar = tid >> 1, ah = (tid & 1) * 16;
  const int br = tid >> 2, bq = (tid & 3) * 8;
  const int m = m0 + ar;
  const int b = m >> 10;
  const int l = m & 1023;
  const int ii = l >> 5;
  const int jj = l & 31;
  const int yy = 2 * ii - 1 + ki;      // pad P=1, stride S=2
  const int xx = 2 * jj - 1 + kj;
  const bool avalid = ((unsigned)yy < (unsigned)NH) && ((unsigned)xx < (unsigned)NW);
  const unsigned short* arow =
      xT + ((size_t)(b * NH + (avalid ? yy : 0)) * NW + (avalid ? xx : 0)) * NC + ah;
  const unsigned short* brow = Wconv + (size_t)kk * NC * NC + (size_t)(n0 + br) * NC + bq;
  if (!avalid) {   // pre-zero halo rows in both buffers once
    u32x4 z = {0u, 0u, 0u, 0u};
    *(u32x4*)&sA[0 * TASZ + ar * 40 + ah]     = z;
    *(u32x4*)&sA[0 * TASZ + ar * 40 + ah + 8] = z;
    *(u32x4*)&sA[1 * TASZ + ar * 40 + ah]     = z;
    *(u32x4*)&sA[1 * TASZ + ar * 40 + ah + 8] = z;
  }
  v8f acc00 = {}, acc01 = {}, acc10 = {}, acc11 = {};
  auto aload = [&](int k0, int buf) {
    if (!avalid) return;
    unsigned short* dst = &sA[buf * TASZ + ar * 40 + ah];
#if USE_ASYNC_LDS
    async_copy16<0>(arow + k0, dst);
    async_copy16<16>(arow + k0, dst);
#else
    *(u32x4*)dst       = *(const u32x4*)(arow + k0);
    *(u32x4*)(dst + 8) = *(const u32x4*)(arow + k0 + 8);
    __builtin_prefetch((const void*)(arow + k0 + 32), 0, 3);
#endif
  };
  auto bload = [&](int k0, int buf) {
    unsigned short* dst = &sB[buf * TBSZ + br * 40 + bq];
#if USE_ASYNC_LDS
    async_copy16<0>(brow + k0, dst);
#else
    *(u32x4*)dst = *(const u32x4*)(brow + k0);
    __builtin_prefetch((const void*)(brow + k0 + 32), 0, 3);
#endif
  };
  gemm_mainloop(sA, sB, aload, bload, lane, wm, wn, acc00, acc01, acc10, acc11);

  const int ln = lane & 15;
  const int rb = (lane < 16) ? 0 : 8;
  const int col0 = n0 + wn + ln;
  const int col1 = col0 + 16;
#pragma unroll
  for (int r = 0; r < 8; ++r) {
    const int row0 = m0 + wm + rb + r;
    const int row1 = row0 + 16;
    vbf[((size_t)row0 * NKK + kk) * NC + col0] = f2bf(acc00[r]);
    vbf[((size_t)row0 * NKK + kk) * NC + col1] = f2bf(acc01[r]);
    vbf[((size_t)row1 * NKK + kk) * NC + col0] = f2bf(acc10[r]);
    vbf[((size_t)row1 * NKK + kk) * NC + col1] = f2bf(acc11[r]);
  }
}

// =====================================================================
// attention apply: o[bl][p][c] = sum_q probs[bl][head(c)][p][q] * v[bl][q][c]
// =====================================================================
__global__ __launch_bounds__(256) void apply_attn_kernel(const float* __restrict__ probs,
                                                         const unsigned short* __restrict__ vbf,
                                                         unsigned short* __restrict__ obf) {
  const int bl = blockIdx.x;
  const float* pb = probs + (size_t)bl * NK4H;
  const unsigned short* vb = vbf + (size_t)bl * NKK * NC;
  unsigned short* ob = obf + (size_t)bl * NKK * NC;
  for (int idx = threadIdx.x; idx < NKK * NC; idx += 256) {
    const int p = idx / NC;
    const int c = idx - p * NC;
    const int head = c >> 5;
    const float* arow = pb + head * 81 + p * 9;
    float s = 0.f;
#pragma unroll
    for (int q = 0; q < 9; ++q) s += arow[q] * bf2f(vb[q * NC + c]);
    ob[idx] = f2bf(s);
  }
}

// =====================================================================
// proj GEMM: A rows built by fused col2im (sum of 1..4 o-rows), B = proj_w [N,K],
// epilogue scatters to NCHW fp32 output with bias
// =====================================================================
__global__ __launch_bounds__(256) void proj_gemm_kernel(const unsigned short* __restrict__ obf,
                                                        const unsigned short* __restrict__ Wproj,
                                                        const float* __restrict__ proj_b,
                                                        float* __restrict__ out) {
  __shared__ __align__(16) unsigned short sA[2 * TASZ];
  __shared__ __align__(16) unsigned short sB[2 * TBSZ];
  const int tid  = threadIdx.x;
  const int lane = tid & 31;
  const int wave = tid >> 5;
  const int wm = (wave & 3) * 32;
  const int wn = (wave >> 2) * 32;
  const int m0 = blockIdx.x * 128;
  const int n0 = blockIdx.y * 64;
  const int ar = tid >> 1, ah = (tid & 1) * 16;
  const int br = tid >> 2, bq = (tid & 3) * 8;
  const int m  = m0 + ar;
  const int b  = m >> 12;
  const int rem = m & 4095;
  const int hh = rem >> 6;
  const int ww = rem & 63;
  int srcRow[4];
  int nsrc = 0;
#pragma unroll
  for (int ki = 0; ki < 3; ++ki) {
    const int t = hh + 1 - ki;
    if (t >= 0 && (t & 1) == 0 && (t >> 1) < PHW) {
      const int ii = t >> 1;
#pragma unroll
      for (int kj = 0; kj < 3; ++kj) {
        const int u = ww + 1 - kj;
        if (u >= 0 && (u & 1) == 0 && (u >> 1) < PHW) {
          const int jj = u >> 1;
          const int bl = (b << 10) + (ii << 5) + jj;
          srcRow[nsrc++] = bl * NKK + (ki * 3 + kj);
        }
      }
    }
  }
  const unsigned short* brow = Wproj + (size_t)(n0 + br) * NC + bq;
  v8f acc00 = {}, acc01 = {}, acc10 = {}, acc11 = {};
  auto aload = [&](int k0, int buf) {   // fused col2im gather-sum, fp32 accum -> bf16
    float acc[16];
#pragma unroll
    for (int e = 0; e < 16; ++e) acc[e] = 0.f;
    for (int s = 0; s < nsrc; ++s) {
      const unsigned short* p = obf + (size_t)srcRow[s] * NC + k0 + ah;
      H8 h0, h1;
      h0.q = *(const u32x4*)(p);
      h1.q = *(const u32x4*)(p + 8);
#pragma unroll
      for (int e = 0; e < 8; ++e) {
        acc[e]     += bf2f(h0.s[e]);
        acc[e + 8] += bf2f(h1.s[e]);
      }
    }
    Frag16 fw;
#pragma unroll
    for (int e = 0; e < 16; ++e) fw.s[e] = f2bf(acc[e]);
    *(u32x4*)&sA[buf * TASZ + ar * 40 + ah]     = fw.q[0];
    *(u32x4*)&sA[buf * TASZ + ar * 40 + ah + 8] = fw.q[1];
  };
  auto bload = [&](int k0, int buf) {
    unsigned short* dst = &sB[buf * TBSZ + br * 40 + bq];
#if USE_ASYNC_LDS
    async_copy16<0>(brow + k0, dst);
#else
    *(u32x4*)dst = *(const u32x4*)(brow + k0);
    __builtin_prefetch((const void*)(brow + k0 + 32), 0, 3);
#endif
  };
  gemm_mainloop(sA, sB, aload, bload, lane, wm, wn, acc00, acc01, acc10, acc11);

  const int ln = lane & 15;
  const int rb = (lane < 16) ? 0 : 8;
  const int col0 = n0 + wn + ln;
  const int col1 = col0 + 16;
  const float bi0 = proj_b[col0];
  const float bi1 = proj_b[col1];
#pragma unroll
  for (int r = 0; r < 8; ++r) {
    const int row0 = m0 + wm + rb + r;
    const int row1 = row0 + 16;
    const int b0i = row0 >> 12, r0 = row0 & 4095;
    const int b1i = row1 >> 12, r1 = row1 & 4095;
    out[(size_t)(b0i * NC + col0) * 4096 + r0] = acc00[r] + bi0;
    out[(size_t)(b0i * NC + col1) * 4096 + r0] = acc01[r] + bi1;
    out[(size_t)(b1i * NC + col0) * 4096 + r1] = acc10[r] + bi0;
    out[(size_t)(b1i * NC + col1) * 4096 + r1] = acc11[r] + bi1;
  }
}

// =====================================================================
extern "C" void kernel_launch(void* const* d_in, const int* in_sizes, int n_in,
                              void* d_out, int out_size, void* d_ws, size_t ws_size,
                              hipStream_t stream) {
  (void)in_sizes; (void)n_in; (void)out_size; (void)ws_size;
  const float* x      = (const float*)d_in[0];
  const float* attn_w = (const float*)d_in[1];
  const float* attn_b = (const float*)d_in[2];
  const float* conv_w = (const float*)d_in[3];
  const float* proj_w = (const float*)d_in[4];
  const float* proj_b = (const float*)d_in[5];
  float* out = (float*)d_out;

  char* ws = (char*)d_ws;
  size_t off = 0;
  auto alloc = [&](size_t bytes) -> char* {
    off = (off + 255) & ~(size_t)255;
    char* p = ws + off;
    off += bytes;
    return p;
  };
  unsigned short* xT     = (unsigned short*)alloc((size_t)NB * NH * NW * NC * 2);
  unsigned short* poolbf = (unsigned short*)alloc((size_t)NBL * NC * 2);
  unsigned short* wattn  = (unsigned short*)alloc((size_t)NK4H * NC * 2);
  unsigned short* wconv  = (unsigned short*)alloc((size_t)NKK * NC * NC * 2);
  unsigned short* wproj  = (unsigned short*)alloc((size_t)NC * NC * 2);
  float*          alog   = (float*)alloc((size_t)NBL * NK4H * 4);
  unsigned short* vbf    = (unsigned short*)alloc((size_t)NBL * NKK * NC * 2);
  unsigned short* obf    = (unsigned short*)alloc((size_t)NBL * NKK * NC * 2);

  cvt_bf16_kernel<<<512,  256, 0, stream>>>(attn_w, wattn, NK4H * NC);
  cvt_bf16_kernel<<<2048, 256, 0, stream>>>(conv_w, wconv, NKK * NC * NC);
  cvt_bf16_kernel<<<256,  256, 0, stream>>>(proj_w, wproj, NC * NC);
  pool_kernel<<<NB * PHW, 256, 0, stream>>>(x, poolbf);
  transpose_kernel<<<dim3(NB * NH, NC / 64), 256, 0, stream>>>(x, xT);
  attn_gemm_kernel<<<dim3(NBL / 128, (NK4H + 63) / 64), 256, 0, stream>>>(poolbf, wattn, attn_b, alog);
  softmax9_kernel<<<(NGROUPS + 255) / 256, 256, 0, stream>>>(alog, NGROUPS);
  conv_gemm_kernel<<<dim3(NBL / 128, NC / 64, NKK), 256, 0, stream>>>(xT, wconv, vbf);
  apply_attn_kernel<<<NBL, 256, 0, stream>>>(alog, vbf, obf);
  proj_gemm_kernel<<<dim3((NB * NH * NW) / 128, NC / 64), 256, 0, stream>>>(obf, wproj, proj_b, out);
}
